// NativeFLABlock_44710609551563
// MI455X (gfx1250) — matrix operand order repaired and checked by reading, not compiled, not run on hardware
//
#include <hip/hip_runtime.h>
#include <hip/hip_bf16.h>
#include <cstdint>
#include <cstddef>

typedef __attribute__((ext_vector_type(16))) _Float16 v16h;
typedef __attribute__((ext_vector_type(8)))  _Float16 v8h;
typedef __attribute__((ext_vector_type(8)))  float    v8f;

#define WMMA_F32_F16(a,b,c) \
  __builtin_amdgcn_wmma_f32_16x16x32_f16(false,(a),false,(b),(short)0,(c),false,false)

namespace {
constexpr int B_   = 8,  T_  = 2048, D_ = 544, H_ = 8, DK_ = 64, DV_ = 128;
constexpr int KDIM = 512, VDIM = 1024, HID = 1632;
constexpr int NT   = B_ * T_;           // 16384 tokens (M of all GEMMs, = 128*128)
constexpr int NP   = 3200;              // fused proj cols padded (actual 3088)
constexpr int QKVC = 2048;              // q(512)+k(512)+v(1024) conv channels
constexpr int NO_PAD  = 640;            // padded 544 (Wo out / Wproj out)
constexpr int NFC_PAD = 1664;           // padded 1632 (fc out / Wproj K)
constexpr int COL_B = 2048, COL_A = 2056, COL_G = 2064;
}

// ---------------------------------------------------------------------------
// WMMA fragment loaders (NT GEMM: A is MxK row-major, B is NxK row-major).
// A 16x32 f16 layout: lane (h=lane>>4, m=lane&15) holds row m,
//   elems 0..7  -> K = k0 + 8h + e        (16B contiguous)
//   elems 8..15 -> K = k0 + 16 + 8h + e   (16B contiguous)
// B 32x16 f16 layout: lane (h, n=lane&15) holds W row n,
//   elems 0..7  -> K = k0 + 8h + e        (16B contiguous)
//   elems 8..15 -> K = k0 + 16 + 8h + e   (16B contiguous)
// ---------------------------------------------------------------------------
__device__ __forceinline__ v16h frag_a(const _Float16* __restrict__ A, int lda,
                                       int k0, int lane) {
  int m = lane & 15, hh = lane >> 4;
  const _Float16* p = A + (size_t)m * lda + k0 + 8 * hh;
  v8h lo = *(const v8h*)p;
  v8h hi = *(const v8h*)(p + 16);
  v16h r;
#pragma unroll
  for (int i = 0; i < 8; ++i) { r[i] = lo[i]; r[i + 8] = hi[i]; }
  return r;
}

__device__ __forceinline__ v16h frag_b(const _Float16* __restrict__ Bw, int ldb,
                                       int k0, int lane) {
  int n = lane & 15, hh = lane >> 4;
  const _Float16* p = Bw + (size_t)n * ldb + k0 + 8 * hh;
  v8h lo = *(const v8h*)p;
  v8h hi = *(const v8h*)(p + 16);
  v16h r;
#pragma unroll
  for (int i = 0; i < 8; ++i) { r[i] = lo[i]; r[i + 8] = hi[i]; }
  return r;
}

// C[M,Npad] = A[M,K] * B[Npad,K]^T.  Block tile 128x128, 8 waves (2 M x 4 N),
// wave tile 64x32 = 4x2 WMMA frags. M, Npad multiples of 128; K multiple of 32.
__global__ __launch_bounds__(256) void gemm_nt(
    const _Float16* __restrict__ A, const _Float16* __restrict__ Bw,
    float* __restrict__ C, int K, int lda, int ldb, int ldc) {
  const int tid  = threadIdx.x;
  const int lane = tid & 31, w = tid >> 5;
  const int wm = w & 1, wn = w >> 1;
  const size_t row0 = (size_t)blockIdx.y * 128 + (size_t)wm * 64;
  const size_t col0 = (size_t)blockIdx.x * 128 + (size_t)wn * 32;
  const _Float16* Ab = A  + row0 * lda;
  const _Float16* Bb = Bw + col0 * ldb;

  v8f acc[4][2];
#pragma unroll
  for (int i = 0; i < 4; ++i)
#pragma unroll
    for (int j = 0; j < 2; ++j)
#pragma unroll
      for (int r = 0; r < 8; ++r) acc[i][j][r] = 0.0f;

  for (int k0 = 0; k0 < K; k0 += 32) {
    v16h a0 = frag_a(Ab,                     lda, k0, lane);
    v16h a1 = frag_a(Ab + (size_t)16 * lda,  lda, k0, lane);
    v16h a2 = frag_a(Ab + (size_t)32 * lda,  lda, k0, lane);
    v16h a3 = frag_a(Ab + (size_t)48 * lda,  lda, k0, lane);
    v16h b0 = frag_b(Bb,                     ldb, k0, lane);
    v16h b1 = frag_b(Bb + (size_t)16 * ldb,  ldb, k0, lane);
    acc[0][0] = WMMA_F32_F16(a0, b0, acc[0][0]);
    acc[0][1] = WMMA_F32_F16(a0, b1, acc[0][1]);
    acc[1][0] = WMMA_F32_F16(a1, b0, acc[1][0]);
    acc[1][1] = WMMA_F32_F16(a1, b1, acc[1][1]);
    acc[2][0] = WMMA_F32_F16(a2, b0, acc[2][0]);
    acc[2][1] = WMMA_F32_F16(a2, b1, acc[2][1]);
    acc[3][0] = WMMA_F32_F16(a3, b0, acc[3][0]);
    acc[3][1] = WMMA_F32_F16(a3, b1, acc[3][1]);
  }

  const int n = lane & 15, hh = lane >> 4;
#pragma unroll
  for (int i = 0; i < 4; ++i)
#pragma unroll
    for (int j = 0; j < 2; ++j) {
      float* Cp = C + (row0 + i * 16 + hh * 8) * ldc + col0 + j * 16 + n;
#pragma unroll
      for (int r = 0; r < 8; ++r) Cp[(size_t)r * ldc] = acc[i][j][r];
    }
}

// ---------------------------------------------------------------------------
// Elementwise / norm / scan kernels
// ---------------------------------------------------------------------------

// (optional residual add) + parameter-free RMSNorm -> f16 activations
__global__ __launch_bounds__(256) void residual_rms(
    const float* __restrict__ x, const float* __restrict__ delta, int ldd,
    const float* __restrict__ svec, float* __restrict__ xm_out,
    _Float16* __restrict__ h16, float eps) {
  const int tok = blockIdx.x, tid = threadIdx.x;
  __shared__ float xs[D_];
  __shared__ float red[256];
  float ss = 0.0f;
  for (int c = tid; c < D_; c += 256) {
    float v = x[(size_t)tok * D_ + c];
    if (delta) v += svec[c] * delta[(size_t)tok * ldd + c];
    xs[c] = v;
    if (xm_out) xm_out[(size_t)tok * D_ + c] = v;
    ss += v * v;
  }
  red[tid] = ss;
  __syncthreads();
  for (int s = 128; s > 0; s >>= 1) {
    if (tid < s) red[tid] += red[tid + s];
    __syncthreads();
  }
  const float scale = rsqrtf(red[0] / (float)D_ + eps);
  for (int c = tid; c < D_; c += 256)
    h16[(size_t)tok * D_ + c] = (_Float16)(xs[c] * scale);
}

// build fused f32->f16 weight [3200 x 544]: rows = Wq|Wk|Wv|Wb|Wa|Wg|pad0
__global__ void build_wc(const float* __restrict__ Wq, const float* __restrict__ Wk,
                         const float* __restrict__ Wv, const float* __restrict__ Wb,
                         const float* __restrict__ Wa, const float* __restrict__ Wg,
                         _Float16* __restrict__ dst) {
  int i = blockIdx.x * blockDim.x + threadIdx.x;
  if (i >= NP * D_) return;
  int nrow = i / D_, k = i - nrow * D_;
  float v = 0.0f;
  if      (nrow < 512)  v = Wq[(size_t)nrow * D_ + k];
  else if (nrow < 1024) v = Wk[(size_t)(nrow - 512) * D_ + k];
  else if (nrow < 2048) v = Wv[(size_t)(nrow - 1024) * D_ + k];
  else if (nrow < 2056) v = Wb[(size_t)(nrow - 2048) * D_ + k];
  else if (nrow < 2064) v = Wa[(size_t)(nrow - 2056) * D_ + k];
  else if (nrow < 3088) v = Wg[(size_t)(nrow - 2064) * D_ + k];
  dst[i] = (_Float16)v;
}

// generic f32->f16 with zero padding in N and K
__global__ void cvt_pad(const float* __restrict__ src, _Float16* __restrict__ dst,
                        int N, int K, int Npad, int Kpad) {
  int i = blockIdx.x * blockDim.x + threadIdx.x;
  if (i >= Npad * Kpad) return;
  int n = i / Kpad, k = i - n * Kpad;
  dst[i] = (n < N && k < K) ? (_Float16)src[(size_t)n * K + k] : (_Float16)0.0f;
}

// causal depthwise conv (K=4) + SiLU for the q|k|v slab of proj
__global__ __launch_bounds__(256) void conv_silu(
    const float* __restrict__ proj, const float* __restrict__ cq,
    const float* __restrict__ ck, const float* __restrict__ cv,
    float* __restrict__ qkv) {
  int idx = blockIdx.x * blockDim.x + threadIdx.x;
  if (idx >= B_ * T_ * QKVC) return;
  int c = idx & (QKVC - 1);
  int t = (idx >> 11) & (T_ - 1);
  int b = idx >> 22;
  const float* w = (c < 512) ? (cq + (size_t)c * 4)
                 : (c < 1536) ? ((c < 1024) ? (ck + (size_t)(c - 512) * 4)
                                            : (cv + (size_t)(c - 1024) * 4))
                              : (cv + (size_t)(c - 1024) * 4);
  float y = 0.0f;
#pragma unroll
  for (int i = 0; i < 4; ++i) {
    int tt = t - 3 + i;
    if (tt >= 0) y += proj[((size_t)(b * T_ + tt)) * NP + c] * w[i];
  }
  qkv[idx] = y / (1.0f + __expf(-y));   // SiLU
}

// beta = 2*sigmoid(.), g = -exp(A_log)*softplus(. + dt_bias)
__global__ void beta_g(const float* __restrict__ proj, const float* __restrict__ A_log,
                       const float* __restrict__ dt_bias, float* __restrict__ bg) {
  int idx = blockIdx.x * blockDim.x + threadIdx.x;
  if (idx >= NT * H_) return;
  int h = idx & 7, tok = idx >> 3;
  float pb = proj[(size_t)tok * NP + COL_B + h];
  float pa = proj[(size_t)tok * NP + COL_A + h];
  float beta = 2.0f / (1.0f + __expf(-pb));
  float z = pa + dt_bias[h];
  float sp = (z > 20.0f) ? z : log1pf(__expf(z));
  bg[(size_t)tok * 16 + h]     = beta;
  bg[(size_t)tok * 16 + 8 + h] = -__expf(A_log[h]) * sp;
}

// L2-normalize q (and * DK^-0.5) and k over head dim 64, in place.
__global__ __launch_bounds__(32) void l2norm_qk(float* __restrict__ qkv) {
  int bid = blockIdx.x;
  int which = bid & 1;          // 0 = q, 1 = k
  int h = (bid >> 1) & 7;
  int tok = bid >> 4;
  float* base = qkv + (size_t)tok * QKVC + (which ? 512 : 0) + h * 64;
  int lane = threadIdx.x;
  float a0 = base[lane], a1 = base[lane + 32];
  float ss = a0 * a0 + a1 * a1;
#pragma unroll
  for (int off = 16; off > 0; off >>= 1) ss += __shfl_xor(ss, off, 32);
  float sc = rsqrtf(ss + 1e-6f) * (which ? 1.0f : 0.125f);  // 0.125 = 64^-0.5
  base[lane]      = a0 * sc;
  base[lane + 32] = a1 * sc;
}

// gated delta-rule recurrence. grid = B*H*2 (v-dim split 2-way), block = 64.
// Lane owns one v-column; S[:,v] (64 deep) lives in registers.
__global__ __launch_bounds__(64) void gdn_scan(const float* __restrict__ qkv,
                                               const float* __restrict__ bg,
                                               float* __restrict__ obuf) {
  const int blk = blockIdx.x;
  const int vs  = blk & 1;
  const int bh  = blk >> 1;
  const int b = bh >> 3, h = bh & 7;
  const int tid = threadIdx.x;
  const int vcol = vs * 64 + tid;
  float S[64];
#pragma unroll
  for (int j = 0; j < 64; ++j) S[j] = 0.0f;
  __shared__ float sk[64];
  __shared__ float sq[64];
  const size_t base = (size_t)b * T_ * QKVC;
  for (int t = 0; t < T_; ++t) {
    const size_t tb  = base + (size_t)t * QKVC;
    const size_t tok = (size_t)(b * T_ + t);
    sk[tid] = qkv[tb + 512 + h * 64 + tid];
    sq[tid] = qkv[tb + h * 64 + tid];
    float vt   = qkv[tb + 1024 + h * 128 + vcol];
    float beta = bg[tok * 16 + h];
    float g    = bg[tok * 16 + 8 + h];
    if (t + 1 < T_) __builtin_prefetch(qkv + tb + QKVC + h * 64, 0, 1);
    __syncthreads();
    const float eg = __expf(g);
    float kv0 = 0, kv1 = 0, kv2 = 0, kv3 = 0;
#pragma unroll
    for (int j = 0; j < 64; j += 4) {
      float s0 = S[j] * eg, s1 = S[j + 1] * eg, s2 = S[j + 2] * eg, s3 = S[j + 3] * eg;
      S[j] = s0; S[j + 1] = s1; S[j + 2] = s2; S[j + 3] = s3;
      kv0 += s0 * sk[j]; kv1 += s1 * sk[j + 1];
      kv2 += s2 * sk[j + 2]; kv3 += s3 * sk[j + 3];
    }
    const float vres = (vt - ((kv0 + kv1) + (kv2 + kv3))) * beta;
    float o0 = 0, o1 = 0, o2 = 0, o3 = 0;
#pragma unroll
    for (int j = 0; j < 64; j += 4) {
      float s0 = S[j] + sk[j] * vres;         S[j] = s0;     o0 += s0 * sq[j];
      float s1 = S[j + 1] + sk[j + 1] * vres; S[j + 1] = s1; o1 += s1 * sq[j + 1];
      float s2 = S[j + 2] + sk[j + 2] * vres; S[j + 2] = s2; o2 += s2 * sq[j + 2];
      float s3 = S[j + 3] + sk[j + 3] * vres; S[j + 3] = s3; o3 += s3 * sq[j + 3];
    }
    obuf[tok * VDIM + h * 128 + vcol] = (o0 + o1) + (o2 + o3);
    __syncthreads();
  }
}

// gated RMSNorm over DV=128: o16 = rms(o)*o_norm_w*silu(gate), f16 out
__global__ __launch_bounds__(128) void gate_norm(
    const float* __restrict__ obuf, const float* __restrict__ proj,
    const float* __restrict__ o_norm_w, _Float16* __restrict__ o16) {
  const int tok = blockIdx.x >> 3, h = blockIdx.x & 7;
  const int tid = threadIdx.x, lane = tid & 31, wv = tid >> 5;
  float ov = obuf[(size_t)tok * VDIM + h * 128 + tid];
  float ss = ov * ov;
#pragma unroll
  for (int off = 16; off > 0; off >>= 1) ss += __shfl_xor(ss, off, 32);
  __shared__ float part[4];
  if (lane == 0) part[wv] = ss;
  __syncthreads();
  float tot = part[0] + part[1] + part[2] + part[3];
  float rn = ov * rsqrtf(tot / (float)DV_ + 1e-5f);
  float gate = proj[(size_t)tok * NP + COL_G + h * 128 + tid];
  float sg = gate / (1.0f + __expf(-gate));
  o16[(size_t)tok * VDIM + h * 128 + tid] = (_Float16)(rn * o_norm_w[tid] * sg);
}

// m16 = f16(relu(m)^2)
__global__ void relu2_f16(const float* __restrict__ m, _Float16* __restrict__ m16,
                          int n) {
  int i = blockIdx.x * blockDim.x + threadIdx.x;
  if (i >= n) return;
  float r = m[i];
  r = (r > 0.0f) ? r : 0.0f;
  m16[i] = (_Float16)(r * r);
}

// out += mlp_scale * dp   (dp strided NO_PAD)
__global__ void final_add(float* __restrict__ out, const float* __restrict__ dp,
                          const float* __restrict__ mscale) {
  int i = blockIdx.x * blockDim.x + threadIdx.x;
  if (i >= NT * D_) return;
  int tok = i / D_, c = i - tok * D_;
  out[i] += mscale[c] * dp[(size_t)tok * NO_PAD + c];
}

// ---------------------------------------------------------------------------
extern "C" void kernel_launch(void* const* d_in, const int* in_sizes, int n_in,
                              void* d_out, int out_size, void* d_ws, size_t ws_size,
                              hipStream_t stream) {
  const float* x       = (const float*)d_in[0];
  const float* Wq      = (const float*)d_in[1];
  const float* Wk      = (const float*)d_in[2];
  const float* Wv      = (const float*)d_in[3];
  const float* Wb      = (const float*)d_in[4];
  const float* Wa      = (const float*)d_in[5];
  const float* Wg      = (const float*)d_in[6];
  const float* Wo      = (const float*)d_in[7];
  const float* cq      = (const float*)d_in[8];
  const float* ck      = (const float*)d_in[9];
  const float* cv      = (const float*)d_in[10];
  const float* A_log   = (const float*)d_in[11];
  const float* dt_bias = (const float*)d_in[12];
  const float* onw     = (const float*)d_in[13];
  const float* Wfc     = (const float*)d_in[14];
  const float* Wpj     = (const float*)d_in[15];
  const float* gdns    = (const float*)d_in[16];
  const float* mlps    = (const float*)d_in[17];
  float* out = (float*)d_out;

  // ---- workspace carve (all sizes are multiples of 256 bytes) ----
  char* p = (char*)d_ws;
  auto carve = [&p](size_t bytes) { char* r = p; p += bytes; return r; };
  _Float16* h16   = (_Float16*)carve((size_t)NT * D_ * 2);        // 17.8 MB
  _Float16* Wc16  = (_Float16*)carve((size_t)NP * D_ * 2);        //  3.5 MB
  _Float16* Wo16  = (_Float16*)carve((size_t)NO_PAD * VDIM * 2);  //  1.3 MB
  _Float16* Wfc16 = (_Float16*)carve((size_t)NFC_PAD * D_ * 2);   //  1.8 MB
  _Float16* Wpj16 = (_Float16*)carve((size_t)NO_PAD * NFC_PAD * 2);// 2.1 MB
  float*    proj  = (float*)carve((size_t)NT * NP * 4);           // 210 MB
  float*    qkv   = (float*)carve((size_t)NT * QKVC * 4);         // 134 MB
  float*    bg    = (float*)carve((size_t)NT * 16 * 4);           //  1.0 MB
  float*    obuf  = (float*)carve((size_t)NT * VDIM * 4);         //  67 MB
  _Float16* o16   = (_Float16*)carve((size_t)NT * VDIM * 2);      //  34 MB
  _Float16* m16   = (_Float16*)carve((size_t)NT * NFC_PAD * 2);   //  55 MB
  // dead-buffer aliases:
  float* mbuf  = proj;   // fc GEMM out  (proj dead after gate_norm)
  float* dxbuf = obuf;   // Wo GEMM out  (obuf dead after gate_norm)
  float* dpbuf = qkv;    // Wproj GEMM out (qkv dead after scan)

  // ---- weight conversion (f32 -> padded f16) ----
  build_wc<<<(NP * D_ + 255) / 256, 256, 0, stream>>>(Wq, Wk, Wv, Wb, Wa, Wg, Wc16);
  cvt_pad<<<(NO_PAD * VDIM + 255) / 256, 256, 0, stream>>>(Wo, Wo16, D_, VDIM, NO_PAD, VDIM);
  cvt_pad<<<(NFC_PAD * D_ + 255) / 256, 256, 0, stream>>>(Wfc, Wfc16, HID, D_, NFC_PAD, D_);
  cvt_pad<<<(NO_PAD * NFC_PAD + 255) / 256, 256, 0, stream>>>(Wpj, Wpj16, D_, HID, NO_PAD, NFC_PAD);

  // ---- h = rmsnorm(x) ----
  residual_rms<<<NT, 256, 0, stream>>>(x, nullptr, 0, nullptr, nullptr, h16, 1e-6f);

  // ---- fused projection GEMM: proj = h @ Wc^T  (M=16384,N=3200,K=544) ----
  gemm_nt<<<dim3(NP / 128, NT / 128), 256, 0, stream>>>(h16, Wc16, proj, D_, D_, D_, NP);

  // ---- conv + silu, beta/g, l2 norm ----
  conv_silu<<<(B_ * T_ * QKVC + 255) / 256, 256, 0, stream>>>(proj, cq, ck, cv, qkv);
  beta_g<<<(NT * H_ + 255) / 256, 256, 0, stream>>>(proj, A_log, dt_bias, bg);
  l2norm_qk<<<NT * H_ * 2, 32, 0, stream>>>(qkv);

  // ---- gated delta-rule scan ----
  gdn_scan<<<B_ * H_ * 2, 64, 0, stream>>>(qkv, bg, obuf);

  // ---- gated RMSNorm -> o16 ----
  gate_norm<<<NT * H_, 128, 0, stream>>>(obuf, proj, onw, o16);

  // ---- dx = o @ Wo^T ; x_mid = x + gdn_scale*dx ; h2 = rms(x_mid) ----
  gemm_nt<<<dim3(NO_PAD / 128, NT / 128), 256, 0, stream>>>(o16, Wo16, dxbuf, VDIM, VDIM, VDIM, NO_PAD);
  residual_rms<<<NT, 256, 0, stream>>>(x, dxbuf, NO_PAD, gdns, out, h16, 1e-6f);

  // ---- MLP: m = relu(h2 @ Wfc^T)^2 ; out += mlp_scale * (m @ Wpj^T) ----
  gemm_nt<<<dim3(NFC_PAD / 128, NT / 128), 256, 0, stream>>>(h16, Wfc16, mbuf, D_, D_, D_, NFC_PAD);
  relu2_f16<<<(NT * NFC_PAD + 255) / 256, 256, 0, stream>>>(mbuf, m16, NT * NFC_PAD);
  gemm_nt<<<dim3(NO_PAD / 128, NT / 128), 256, 0, stream>>>(m16, Wpj16, dpbuf, NFC_PAD, NFC_PAD, NFC_PAD, NO_PAD);
  final_add<<<(NT * D_ + 255) / 256, 256, 0, stream>>>(out, dpbuf, mlps);

  (void)in_sizes; (void)n_in; (void)out_size; (void)ws_size;
}
